// WukongExpert_24764781429079
// MI455X (gfx1250) — compile-verified, matrix operand order checked by reading, not run on hardware
//
#include <hip/hip_runtime.h>

typedef __attribute__((ext_vector_type(16))) _Float16 v16h;
typedef __attribute__((ext_vector_type(4)))  _Float16 v4h;
typedef __attribute__((ext_vector_type(8)))  float    v8f;
typedef __attribute__((ext_vector_type(4)))  unsigned int v4u;
typedef __attribute__((ext_vector_type(8)))  int      v8i;
typedef __attribute__((ext_vector_type(4)))  int      v4i;

#define NB 4096  // batch
#define NT 64    // tokens
#define ND 128   // d_model
#define NK 8192  // T*D

#if defined(__has_builtin)
#if __has_builtin(__builtin_amdgcn_tensor_load_to_lds) && \
    __has_builtin(__builtin_amdgcn_s_wait_tensorcnt)
#define HAVE_TDM 1
#endif
#endif

__device__ __forceinline__ v8f wmma_f16(v16h a, v16h b, v8f c) {
  return __builtin_amdgcn_wmma_f32_16x16x32_f16(false, a, false, b, (short)0, c,
                                                false, false);
}

// 16x32 f16 A/B fragment from an LDS tile stored [row][k] (ISA 7.12.2 layout):
// lane l -> row m=l&15, group g=l>>4; VGPR v<4: K=8g+2v,+1 ; v>=4: K=16+8g+2(v-4),+1.
__device__ __forceinline__ v16h frag_ld_lds(const _Float16* p, int stride, int lane) {
  const int m = lane & 15, g = lane >> 4;
  const _Float16* row = p + m * stride + g * 8;
  v16h f;
#pragma unroll
  for (int v = 0; v < 8; ++v) {
    const int off = (v < 4) ? (2 * v) : (2 * v + 8);
    f[2 * v]     = row[off];
    f[2 * v + 1] = row[off + 1];
  }
  return f;
}

// Same gather from a row-major f32 global matrix (row = frag column n),
// K-pairs loaded as float2 (b64) with inline f32->f16 convert.
__device__ __forceinline__ v16h frag_ld_global_f32(const float* p, int stride, int lane) {
  const int n = lane & 15, g = lane >> 4;
  const float* row = p + (size_t)n * stride + g * 8;
  v16h f;
#pragma unroll
  for (int v = 0; v < 8; ++v) {
    const int off = (v < 4) ? (2 * v) : (2 * v + 8);
    const float2 pr = *(const float2*)&row[off];
    f[2 * v]     = (_Float16)pr.x;
    f[2 * v + 1] = (_Float16)pr.y;
  }
  return f;
}

// ---------------------------------------------------------------------------
// Kernel 1: H[B,128] = relu(X[B,8192] @ W1^T + b1), H stored f16.
// 256 thr / 8 waves, M-tile 32, K-tile 32, register double-buffered staging.
// ---------------------------------------------------------------------------
#define K1_MT 32
#define K1_KT 32
#define K1_AS (K1_KT + 8)
#define K1_BS (K1_KT + 8)

__global__ void __launch_bounds__(256)
k1_mlp1(const float* __restrict__ X, const float* __restrict__ W1,
        const float* __restrict__ b1, _Float16* __restrict__ H) {
  __shared__ _Float16 As[K1_MT * K1_AS];  // [m][k]
  __shared__ _Float16 Bs[ND * K1_BS];     // [n][k]
  const int tid = threadIdx.x, wave = tid >> 5, lane = tid & 31;
  const int m0 = blockIdx.x * K1_MT;
  const int mt = wave & 1;   // 2 m-tiles of 16
  const int ng = wave >> 1;  // 4 n-groups x 2 n-tiles = 128 cols
  v8f acc0 = {}, acc1 = {};

  // per-thread staging coordinates (float4 granularity)
  const int ar = tid >> 3, ac = (tid & 7) * 4;  // A: 32x32
  float4 aR;
  float4 bR[4];

  auto ld_tile = [&](int k0) {
    aR = *(const float4*)&X[(size_t)(m0 + ar) * NK + k0 + ac];
#pragma unroll
    for (int i = 0; i < 4; ++i) {  // B: 128x32
      const int e = tid * 4 + i * 1024;
      const int n = e >> 5, c = e & 31;
      bR[i] = *(const float4*)&W1[(size_t)n * NK + k0 + c];
    }
  };
  auto st_tile = [&]() {
    v4h h;
    h[0] = (_Float16)aR.x; h[1] = (_Float16)aR.y;
    h[2] = (_Float16)aR.z; h[3] = (_Float16)aR.w;
    *(v4h*)&As[ar * K1_AS + ac] = h;
#pragma unroll
    for (int i = 0; i < 4; ++i) {
      const int e = tid * 4 + i * 1024;
      const int n = e >> 5, c = e & 31;
      v4h hb;
      hb[0] = (_Float16)bR[i].x; hb[1] = (_Float16)bR[i].y;
      hb[2] = (_Float16)bR[i].z; hb[3] = (_Float16)bR[i].w;
      *(v4h*)&Bs[n * K1_BS + c] = hb;
    }
  };

  ld_tile(0);
  for (int k0 = 0; k0 < NK; k0 += K1_KT) {
    __syncthreads();
    st_tile();
    __syncthreads();
    if (k0 + K1_KT < NK) ld_tile(k0 + K1_KT);  // prefetch next tile (overlaps WMMA)

    const v16h a  = frag_ld_lds(As + mt * 16 * K1_AS, K1_AS, lane);
    const v16h f0 = frag_ld_lds(Bs + (ng * 2 + 0) * 16 * K1_BS, K1_BS, lane);
    const v16h f1 = frag_ld_lds(Bs + (ng * 2 + 1) * 16 * K1_BS, K1_BS, lane);
    acc0 = wmma_f16(a, f0, acc0);
    acc1 = wmma_f16(a, f1, acc1);
  }

  const int g = lane >> 4, nl = lane & 15;
#pragma unroll
  for (int r = 0; r < 8; ++r) {
    const int m = m0 + mt * 16 + r + 8 * g;
    int n = (ng * 2 + 0) * 16 + nl;
    float v = acc0[r] + b1[n];
    H[(size_t)m * ND + n] = (_Float16)(v > 0.f ? v : 0.f);
    n = (ng * 2 + 1) * 16 + nl;
    v = acc1[r] + b1[n];
    H[(size_t)m * ND + n] = (_Float16)(v > 0.f ? v : 0.f);
  }
}

// ---------------------------------------------------------------------------
// Kernel 2: Y[B,8192] = X + H @ W2^T + b2  (residual pre-folded)
// M-tile 16 batches, N-tile 512. A (H tile, f16) staged via TDM
// tensor_load_to_lds with HW row padding (64 DW rows + 4 DW pad == ND+8 stride).
// ---------------------------------------------------------------------------
__global__ void __launch_bounds__(256)
k2_mlp2(const _Float16* __restrict__ H, const float* __restrict__ W2,
        const float* __restrict__ b2, const float* __restrict__ X,
        float* __restrict__ Y) {
  __shared__ _Float16 As[16 * (ND + 8)];
  const int AS = ND + 8;
  const int tid = threadIdx.x, wave = tid >> 5, lane = tid & 31;
  const int m0 = blockIdx.x * 16;
  const int n0 = blockIdx.y * 512 + wave * 64;

#ifdef HAVE_TDM
  if (wave == 0) {
    const unsigned long long ga =
        (unsigned long long)(uintptr_t)(const void*)(H + (size_t)m0 * ND);
    const unsigned int lds = (unsigned int)(uintptr_t)(void*)&As[0];
    v4u g0;
    g0[0] = 1u;  // count=1 user descriptor
    g0[1] = lds;
    g0[2] = (unsigned int)ga;
    g0[3] = (unsigned int)((ga >> 32) & 0x01FFFFFFu) | (2u << 30);  // type=2
    v8i g1;
    // pad_amount=4DW(code 3), pad_interval=64DW(code 5), pad_en, data_size=2B
    g1[0] = (int)((3u << 25) | (5u << 22) | (1u << 20) | (1u << 16));
    g1[1] = (int)(((unsigned)ND & 0xFFFFu) << 16);   // tensor_dim0.lo16
    g1[2] = (int)(((unsigned)NB & 0xFFFFu) << 16);   // dim0.hi16=0 | tensor_dim1.lo16
    g1[3] = (int)(((unsigned)ND & 0xFFFFu) << 16);   // dim1.hi16=0 | tile_dim0=128
    g1[4] = 16;                                      // tile_dim1=16, tile_dim2=0
    g1[5] = ND;                                      // tensor_dim0_stride=128
    g1[6] = 0;
    g1[7] = 0;
    v4i gz;
    gz[0] = gz[1] = gz[2] = gz[3] = 0;
    v8i gz8;
    gz8[0] = gz8[1] = gz8[2] = gz8[3] = 0;
    gz8[4] = gz8[5] = gz8[6] = gz8[7] = 0;
    // 6-arg form (clang-23 / therock-10.0 headers)
    __builtin_amdgcn_tensor_load_to_lds(g0, g1, gz, gz, gz8, 0);
    __builtin_amdgcn_s_wait_tensorcnt(0);
  }
  __syncthreads();
#else
#pragma unroll
  for (int i = 0; i < 8; ++i) {  // 16x128 f16, 8/thread
    const int idx = tid + i * 256;
    const int r = idx >> 7, c = idx & 127;
    As[r * AS + c] = H[(size_t)(m0 + r) * ND + c];
  }
  __syncthreads();
#endif

  v8f acc[4] = {v8f{}, v8f{}, v8f{}, v8f{}};
#pragma unroll
  for (int ks = 0; ks < 4; ++ks) {
    const v16h a = frag_ld_lds(As + ks * 32, AS, lane);
#pragma unroll
    for (int nt = 0; nt < 4; ++nt) {
      const v16h b = frag_ld_global_f32(
          W2 + (size_t)(n0 + nt * 16) * ND + ks * 32, ND, lane);
      acc[nt] = wmma_f16(a, b, acc[nt]);
    }
  }

  const int g = lane >> 4, nl = lane & 15;
#pragma unroll
  for (int nt = 0; nt < 4; ++nt) {
    const int n = n0 + nt * 16 + nl;
    const float bias = b2[n];
#pragma unroll
    for (int r = 0; r < 8; ++r) {
      const int m = m0 + r + 8 * g;
      const size_t o = (size_t)m * NK + n;
      Y[o] = acc[nt][r] + bias + X[o];
    }
  }
}

// ---------------------------------------------------------------------------
// Kernel 3: per-batch attention + residual + LayerNorm. 1 WG (256 thr) / batch.
// ---------------------------------------------------------------------------
__global__ void __launch_bounds__(256)
k3_attn_ln(const float* __restrict__ X, const float* __restrict__ gamma,
           const float* __restrict__ beta, float* __restrict__ Y) {
  __shared__ _Float16 Xs[NT * (ND + 8)];  // X_b   [t][d]
  __shared__ _Float16 Xt[ND * (NT + 8)];  // X_b^T [d][t]
  __shared__ float    Sc[NT * (NT + 4)];  // scores f32
  __shared__ _Float16 At[NT * (NT + 8)];  // attn f16 [t][s]
  __shared__ float    Yd[NT * (ND + 4)];  // X_dot f32, then y
  const int XS = ND + 8, XTS = NT + 8, SS = NT + 4, ATS = NT + 8, YS = ND + 4;
  const int tid = threadIdx.x, wave = tid >> 5, lane = tid & 31;
  const int bidx = blockIdx.x;
  const float* Xb = X + (size_t)bidx * NK;

#pragma unroll
  for (int i = 0; i < 8; ++i) {  // 8192 f32 as float4, both layouts
    const int e = tid * 4 + i * 1024;
    const int t = e >> 7, d = e & 127;
    const float4 v = *(const float4*)&Xb[e];
    v4h h;
    h[0] = (_Float16)v.x; h[1] = (_Float16)v.y;
    h[2] = (_Float16)v.z; h[3] = (_Float16)v.w;
    *(v4h*)&Xs[t * XS + d] = h;
    Xt[(d + 0) * XTS + t] = h[0];
    Xt[(d + 1) * XTS + t] = h[1];
    Xt[(d + 2) * XTS + t] = h[2];
    Xt[(d + 3) * XTS + t] = h[3];
  }
  __syncthreads();

  {  // scores = X X^T / sqrt(D): 4x4 tiles, 2 per wave, K=128
    const int mt = wave >> 1, ntb = (wave & 1) * 2;
    v8f s0 = {}, s1 = {};
#pragma unroll
    for (int ks = 0; ks < 4; ++ks) {
      const v16h a  = frag_ld_lds(Xs + mt * 16 * XS + ks * 32, XS, lane);
      const v16h f0 = frag_ld_lds(Xs + (ntb + 0) * 16 * XS + ks * 32, XS, lane);
      const v16h f1 = frag_ld_lds(Xs + (ntb + 1) * 16 * XS + ks * 32, XS, lane);
      s0 = wmma_f16(a, f0, s0);
      s1 = wmma_f16(a, f1, s1);
    }
    const float scale = 0.08838834764831845f;  // 1/sqrt(128)
    const int g = lane >> 4, nl = lane & 15;
#pragma unroll
    for (int r = 0; r < 8; ++r) {
      const int t = mt * 16 + r + 8 * g;
      Sc[t * SS + (ntb + 0) * 16 + nl] = s0[r] * scale;
      Sc[t * SS + (ntb + 1) * 16 + nl] = s1[r] * scale;
    }
  }
  __syncthreads();

  if (tid < NT) {  // f32 softmax, one row per thread
    float mx = -3.0e38f;
#pragma unroll
    for (int s = 0; s < NT; ++s) mx = fmaxf(mx, Sc[tid * SS + s]);
    float sum = 0.f;
#pragma unroll
    for (int s = 0; s < NT; ++s) sum += __expf(Sc[tid * SS + s] - mx);
    const float inv = 1.f / sum;
#pragma unroll
    for (int s = 0; s < NT; ++s)
      At[tid * ATS + s] = (_Float16)(__expf(Sc[tid * SS + s] - mx) * inv);
  }
  __syncthreads();

  {  // X_dot = attn @ X_b : 4x8 tiles, 4 per wave, K=64
    const int mt = wave >> 1, ntb = (wave & 1) * 4;
    v8f acc[4] = {v8f{}, v8f{}, v8f{}, v8f{}};
#pragma unroll
    for (int ks = 0; ks < 2; ++ks) {
      const v16h a = frag_ld_lds(At + mt * 16 * ATS + ks * 32, ATS, lane);
#pragma unroll
      for (int nt = 0; nt < 4; ++nt) {
        const v16h b = frag_ld_lds(Xt + (ntb + nt) * 16 * XTS + ks * 32, XTS, lane);
        acc[nt] = wmma_f16(a, b, acc[nt]);
      }
    }
    const int g = lane >> 4, nl = lane & 15;
#pragma unroll
    for (int nt = 0; nt < 4; ++nt)
#pragma unroll
      for (int r = 0; r < 8; ++r) {
        const int t = mt * 16 + r + 8 * g;
        Yd[t * YS + (ntb + nt) * 16 + nl] = acc[nt][r];
      }
  }
  __syncthreads();

  {  // y = Y(=X+X_deep) + X_dot ; LayerNorm over D. 4 threads per row.
    const int t = tid >> 2, q = tid & 3;
    const size_t base = (size_t)bidx * NK + (size_t)t * ND;
    float s1 = 0.f, s2 = 0.f;
#pragma unroll
    for (int i = 0; i < 8; ++i) {
      const int d = q * 32 + i * 4;
      const float4 yv = *(const float4*)&Y[base + d];
      float4 lv = *(const float4*)&Yd[t * YS + d];
      lv.x += yv.x; lv.y += yv.y; lv.z += yv.z; lv.w += yv.w;
      *(float4*)&Yd[t * YS + d] = lv;
      s1 += lv.x + lv.y + lv.z + lv.w;
      s2 += lv.x * lv.x + lv.y * lv.y + lv.z * lv.z + lv.w * lv.w;
    }
    s1 += __shfl_xor(s1, 1); s2 += __shfl_xor(s2, 1);
    s1 += __shfl_xor(s1, 2); s2 += __shfl_xor(s2, 2);
    const float mean = s1 * (1.f / 128.f);
    const float var  = s2 * (1.f / 128.f) - mean * mean;
    const float inv  = rsqrtf(var + 1e-5f);
#pragma unroll
    for (int i = 0; i < 8; ++i) {
      const int d = q * 32 + i * 4;
      const float4 lv = *(const float4*)&Yd[t * YS + d];
      const float4 gv = *(const float4*)&gamma[d];
      const float4 bv = *(const float4*)&beta[d];
      float4 o;
      o.x = (lv.x - mean) * inv * gv.x + bv.x;
      o.y = (lv.y - mean) * inv * gv.y + bv.y;
      o.z = (lv.z - mean) * inv * gv.z + bv.z;
      o.w = (lv.w - mean) * inv * gv.w + bv.w;
      *(float4*)&Y[base + d] = o;
    }
  }
}

extern "C" void kernel_launch(void* const* d_in, const int* in_sizes, int n_in,
                              void* d_out, int out_size, void* d_ws, size_t ws_size,
                              hipStream_t stream) {
  const float* X     = (const float*)d_in[0];
  const float* W1    = (const float*)d_in[1];
  const float* b1    = (const float*)d_in[2];
  const float* W2    = (const float*)d_in[3];
  const float* b2    = (const float*)d_in[4];
  const float* gamma = (const float*)d_in[5];
  const float* beta  = (const float*)d_in[6];
  float* Y = (float*)d_out;
  _Float16* H = (_Float16*)d_ws;  // 4096*128*2 = 1 MB scratch

  k1_mlp1<<<NB / K1_MT, 256, 0, stream>>>(X, W1, b1, H);
  k2_mlp2<<<dim3(NB / 16, NK / 512), 256, 0, stream>>>(H, W2, b2, X, Y);
  k3_attn_ln<<<NB, 256, 0, stream>>>(X, gamma, beta, Y);
}
